// SMNetEncoder_23579370455326
// MI455X (gfx1250) — matrix operand order repaired and checked by reading, not compiled, not run on hardware
//
#include <hip/hip_runtime.h>

// ---------------------------------------------------------------------------
// GINEConv x3 encoder for MI455X (gfx1250, wave32, WMMA bf16 16x16x32).
// Memory-bound: x (25.6MB) lives in 192MB L2; edge_vector stream (205MB/layer)
// dominates HBM -> fuse GEMMs into gather/scatter; LDS tiles stored
// pre-swizzled in WMMA fragment order so fragment loads are ds_load_b128;
// 32-bit element offsets so gathers/atomics use saddr+voffset addressing.
// ---------------------------------------------------------------------------

#define N_NODES_C  50000
#define N_EDGES_C  800000
#define CH         128
#define EDGE_DIM_C 64
#define N_LAYERS_C 3

typedef __attribute__((ext_vector_type(16))) __bf16 v16bf;
typedef __attribute__((ext_vector_type(8)))  float  v8f;

// fp32 -> bf16 round-to-nearest-even
static __device__ inline __bf16 f2bf(float f) {
  unsigned u = __float_as_uint(f);
  unsigned r = (u + 0x7FFFu + ((u >> 16) & 1u)) >> 16;
  unsigned short s = (unsigned short)r;
  return __builtin_bit_cast(__bf16, s);
}

// Swizzled element offset (bf16 units) for A-fragment storage (16 x K tile,
// 32-wide k-tiles). ISA 16-bit A 16x32 layout: lane m+16*((k>>3)&1),
// slot (k&7) + 8*(k&16 ? 1 : 0). Each lane's 16 elems contiguous (32B).
static __device__ inline int a_swz(int m, int k) {
  int kt   = k >> 5;
  int kl   = k & 31;
  int lane = m + (((kl >> 3) & 1) << 4);
  int idx  = (kl & 7) + ((kl & 16) ? 8 : 0);
  return (kt * 32 + lane) * 16 + idx;
}

// Swizzled element offset for B-fragment storage (K x N weights, 32x16 tiles).
// ISA B 32x16 layout: lane (n&15)+16*((k>>3)&1), slot (k&7) + 8*(k&16 ? 1 : 0).
static __device__ inline int b_swz(int k, int n, int nTiles) {
  int kt   = k >> 5;
  int kl   = k & 31;
  int nt   = n >> 4;
  int lane = (n & 15) + (((kl >> 3) & 1) << 4);
  int idx  = (kl & 7) + ((kl & 16) ? 8 : 0);
  return ((kt * nTiles + nt) * 32 + lane) * 16 + idx;
}

__global__ __launch_bounds__(256) void zero_kernel(float* __restrict__ p, long long n) {
  long long i = (long long)blockIdx.x * blockDim.x + threadIdx.x;
  const long long stride = (long long)gridDim.x * blockDim.x;
  for (; i < n; i += stride) p[i] = 0.0f;
}

// ---------------------------------------------------------------------------
// Edge phase: e = ev @ edge_W + edge_b ; msg = relu(x[src] + e) ;
// agg[dst] += msg  (fp32 atomics, L2 resident).
// Block = 256 threads = 8 waves; wave w owns channel tile [16w, 16w+16).
// ---------------------------------------------------------------------------
__global__ __launch_bounds__(256) void edge_gine_kernel(
    const float* __restrict__ x,       // [N, CH]
    const int*   __restrict__ adj,     // [2, E]
    const float* __restrict__ ev,      // [E, EDGE_DIM]
    const float* __restrict__ edge_W,  // [EDGE_DIM, CH] layer slice
    const float* __restrict__ edge_b,  // [CH]
    float*       __restrict__ agg)     // [N, CH] pre-zeroed
{
  __shared__ v16bf Wswz[(EDGE_DIM_C / 32) * (CH / 16) * 32];  // 16 KB
  __shared__ v16bf Eswz[(EDGE_DIM_C / 32) * 32];              // 2 KB
  __shared__ int srcIdx[16], dstIdx[16];                      // element offsets

  __bf16* Wb = (__bf16*)Wswz;
  __bf16* Eb = (__bf16*)Eswz;

  const int tid  = threadIdx.x;
  const int lane = tid & 31;
  const int wave = tid >> 5;               // n-tile 0..7

  // weights -> LDS, bf16, fragment-swizzled (amortized over ~50 tiles/block)
  for (int i = tid; i < EDGE_DIM_C * CH; i += 256) {
    int k = i >> 7, n = i & (CH - 1);
    Wb[b_swz(k, n, CH / 16)] = f2bf(edge_W[i]);
  }
  __syncthreads();

  const int ncol  = wave * 16 + (lane & 15);
  const int mbase = (lane < 16) ? 0 : 8;
  const float bias = edge_b[ncol];

  const int nTiles = (N_EDGES_C + 15) / 16;
  for (int t = blockIdx.x; t < nTiles; t += gridDim.x) {
    const int e0 = t * 16;

    // prefetch next tile's edge features (32 x 128B cachelines = 4KB tile)
    if (tid < 32) {
      long long tn = t + (long long)gridDim.x;
      if (tn < nTiles)
        __builtin_prefetch(ev + tn * 16 * EDGE_DIM_C + tid * 32, 0, 1);
    }

    // edge-feature tile -> LDS, bf16, A-swizzled
    for (int i = tid; i < 16 * EDGE_DIM_C; i += 256) {
      int er = i >> 6, ec = i & 63;
      int e = e0 + er;
      Eb[a_swz(er, ec)] = f2bf(e < N_EDGES_C ? ev[(long long)e * EDGE_DIM_C + ec] : 0.0f);
    }
    if (tid < 16) {
      int e = e0 + tid;
      srcIdx[tid] = (e < N_EDGES_C) ? adj[e] * CH : 0;
      dstIdx[tid] = (e < N_EDGES_C) ? adj[N_EDGES_C + e] * CH : 0;
    }
    __syncthreads();

    v8f acc;
#pragma unroll
    for (int r = 0; r < 8; ++r) acc[r] = bias;

#pragma unroll
    for (int kt = 0; kt < EDGE_DIM_C / 32; ++kt) {
      v16bf a = Eswz[kt * 32 + lane];                              // 2x ds_load_b128
      v16bf b = Wswz[(kt * (CH / 16) + wave) * 32 + lane];         // 2x ds_load_b128
      acc = __builtin_amdgcn_wmma_f32_16x16x32_bf16(false, a, false, b,
                                                    (short)0, acc, false, false);
    }

    // gather phase: issue all 8 loads before consuming (latency overlap)
    float gx[8];
#pragma unroll
    for (int r = 0; r < 8; ++r)
      gx[r] = x[srcIdx[mbase + r] + ncol];

    if (e0 + 16 <= N_EDGES_C) {          // full tile (block-uniform fast path)
#pragma unroll
      for (int r = 0; r < 8; ++r) {
        float v = fmaxf(acc[r] + gx[r], 0.0f);
        atomicAdd(&agg[dstIdx[mbase + r] + ncol], v);
      }
    } else {                             // guarded tail
#pragma unroll
      for (int r = 0; r < 8; ++r) {
        if (e0 + mbase + r < N_EDGES_C) {
          float v = fmaxf(acc[r] + gx[r], 0.0f);
          atomicAdd(&agg[dstIdx[mbase + r] + ncol], v);
        }
      }
    }
    __syncthreads();
  }
}

// ---------------------------------------------------------------------------
// Node phase: z=(1+eps)x+agg -> GEMM1 -> BN -> ReLU -> GEMM2 -> +b2
//             -> LayerNorm -> ReLU -> out.  16-node tiles.
// ---------------------------------------------------------------------------
__global__ __launch_bounds__(256) void node_mlp_kernel(
    const float* __restrict__ x, const float* __restrict__ agg,
    const float* __restrict__ eps_p, int layer,
    const float* __restrict__ W1, const float* __restrict__ b1,
    const float* __restrict__ bn_g, const float* __restrict__ bn_b,
    const float* __restrict__ bn_mean, const float* __restrict__ bn_var,
    const float* __restrict__ W2, const float* __restrict__ b2,
    const float* __restrict__ ln_g, const float* __restrict__ ln_b,
    float* __restrict__ out)
{
  __shared__ v16bf W1swz[(CH / 32) * (CH / 16) * 32];   // 32 KB
  __shared__ v16bf W2swz[(CH / 32) * (CH / 16) * 32];   // 32 KB
  __shared__ v16bf Aswz[(CH / 32) * 32];                // 4 KB (z tile)
  __shared__ v16bf Hswz[(CH / 32) * 32];                // 4 KB (hidden tile)
  __shared__ float Yl[16 * CH];                         // 8 KB (pre-LN, fp32)
  __shared__ float red[16][16][2];
  __shared__ float stats[16][2];

  __bf16* W1b = (__bf16*)W1swz;
  __bf16* W2b = (__bf16*)W2swz;
  __bf16* Ab  = (__bf16*)Aswz;
  __bf16* Hb  = (__bf16*)Hswz;

  const int tid  = threadIdx.x;
  const int lane = tid & 31;
  const int wave = tid >> 5;

  for (int i = tid; i < CH * CH; i += 256) {
    int k = i >> 7, n = i & (CH - 1);
    int o = b_swz(k, n, CH / 16);
    W1b[o] = f2bf(W1[i]);
    W2b[o] = f2bf(W2[i]);
  }
  const float epsk = 1.0f + eps_p[layer];
  __syncthreads();

  const int ncol  = wave * 16 + (lane & 15);
  const int mbase = (lane < 16) ? 0 : 8;

  const int nTiles = (N_NODES_C + 15) / 16;
  for (int t = blockIdx.x; t < nTiles; t += gridDim.x) {
    const int n0 = t * 16;

    // z tile (fp32 compute, bf16 A-swizzled store); 32-bit element offsets
    for (int i = tid; i < 16 * CH; i += 256) {
      int r = i >> 7, c = i & (CH - 1);
      int node = n0 + r;
      float z = 0.0f;
      if (node < N_NODES_C) {
        int idx = node * CH + c;
        z = epsk * x[idx] + agg[idx];
      }
      Ab[a_swz(r, c)] = f2bf(z);
    }
    __syncthreads();

    // GEMM1 + bias + BatchNorm(eval) + ReLU -> H (A-swizzled for GEMM2)
    {
      v8f acc = {};
#pragma unroll
      for (int kt = 0; kt < CH / 32; ++kt) {
        v16bf a = Aswz[kt * 32 + lane];
        v16bf b = W1swz[(kt * (CH / 16) + wave) * 32 + lane];
        acc = __builtin_amdgcn_wmma_f32_16x16x32_bf16(false, a, false, b,
                                                      (short)0, acc, false, false);
      }
      float bb = b1[ncol];
      float sc = bn_g[ncol] * rsqrtf(bn_var[ncol] + 1e-5f);
      float sh = bn_b[ncol] - bn_mean[ncol] * sc;
#pragma unroll
      for (int r = 0; r < 8; ++r) {
        float h = (acc[r] + bb) * sc + sh;
        Hb[a_swz(mbase + r, ncol)] = f2bf(fmaxf(h, 0.0f));
      }
    }
    __syncthreads();

    // GEMM2 + bias -> Yl (fp32)
    {
      v8f acc = {};
#pragma unroll
      for (int kt = 0; kt < CH / 32; ++kt) {
        v16bf a = Hswz[kt * 32 + lane];
        v16bf b = W2swz[(kt * (CH / 16) + wave) * 32 + lane];
        acc = __builtin_amdgcn_wmma_f32_16x16x32_bf16(false, a, false, b,
                                                      (short)0, acc, false, false);
      }
      float bb = b2[ncol];
#pragma unroll
      for (int r = 0; r < 8; ++r)
        Yl[(mbase + r) * CH + ncol] = acc[r] + bb;
    }
    __syncthreads();

    // LayerNorm + ReLU (fp32), 16 threads per row
    {
      const int row = tid >> 4;
      const int sub = tid & 15;
      float s = 0.0f, s2 = 0.0f;
#pragma unroll
      for (int j = 0; j < 8; ++j) {
        float v = Yl[row * CH + sub * 8 + j];
        s += v; s2 += v * v;
      }
      red[row][sub][0] = s;
      red[row][sub][1] = s2;
      __syncthreads();
      if (sub == 0) {
        float ts = 0.0f, ts2 = 0.0f;
#pragma unroll
        for (int j = 0; j < 16; ++j) { ts += red[row][j][0]; ts2 += red[row][j][1]; }
        float mu  = ts * (1.0f / CH);
        float var = ts2 * (1.0f / CH) - mu * mu;
        stats[row][0] = mu;
        stats[row][1] = rsqrtf(var + 1e-5f);
      }
      __syncthreads();
      int node = n0 + row;
      if (node < N_NODES_C) {
        float mu = stats[row][0], rs = stats[row][1];
#pragma unroll
        for (int j = 0; j < 8; ++j) {
          int c = sub * 8 + j;
          float v = (Yl[row * CH + c] - mu) * rs * ln_g[c] + ln_b[c];
          out[node * CH + c] = fmaxf(v, 0.0f);
        }
      }
    }
    __syncthreads();
  }
}

// ---------------------------------------------------------------------------
extern "C" void kernel_launch(void* const* d_in, const int* in_sizes, int n_in,
                              void* d_out, int out_size, void* d_ws, size_t ws_size,
                              hipStream_t stream) {
  const float* feat    = (const float*)d_in[0];
  const int*   adj     = (const int*)  d_in[1];   // [2, E] int32
  const float* ev      = (const float*)d_in[2];
  const float* eps     = (const float*)d_in[3];
  const float* edge_W  = (const float*)d_in[4];
  const float* edge_b  = (const float*)d_in[5];
  const float* W1      = (const float*)d_in[6];
  const float* b1      = (const float*)d_in[7];
  const float* bn_g    = (const float*)d_in[8];
  const float* bn_b    = (const float*)d_in[9];
  const float* bn_mean = (const float*)d_in[10];
  const float* bn_var  = (const float*)d_in[11];
  const float* W2      = (const float*)d_in[12];
  const float* b2      = (const float*)d_in[13];
  const float* ln_g    = (const float*)d_in[14];
  const float* ln_b    = (const float*)d_in[15];

  float* out = (float*)d_out;
  const long long NC = (long long)N_NODES_C * CH;
  // workspace: agg + two ping-pong node buffers (3 * 25.6 MB)
  float* aggb = (float*)d_ws;
  float* bufA = aggb + NC;
  float* bufB = bufA + NC;

  const float* xin = feat;
  for (int k = 0; k < N_LAYERS_C; ++k) {
    float* xout = (k == N_LAYERS_C - 1) ? out : ((k == 0) ? bufA : bufB);

    zero_kernel<<<512, 256, 0, stream>>>(aggb, NC);
    edge_gine_kernel<<<1024, 256, 0, stream>>>(
        xin, adj, ev,
        edge_W + (long long)k * EDGE_DIM_C * CH, edge_b + k * CH, aggb);
    node_mlp_kernel<<<512, 256, 0, stream>>>(
        xin, aggb, eps, k,
        W1 + (long long)k * CH * CH, b1 + k * CH,
        bn_g + k * CH, bn_b + k * CH, bn_mean + k * CH, bn_var + k * CH,
        W2 + (long long)k * CH * CH, b2 + k * CH,
        ln_g, ln_b, xout);

    xin = xout;
  }
}
